// BipartiteGNN_40235253629274
// MI455X (gfx1250) — matrix operand order repaired and checked by reading, not compiled, not run on hardware
//
// CDNA5 / gfx1250 bipartite GNN: bf16 WMMA GEMMs, 64-row tiles for 4x B-fragment
// register reuse, fused gather + segment-sum atomics.
#include <hip/hip_runtime.h>
#include <cstdint>
#include <cstddef>

#define DIM 256
#define GNUM 64
#define LSTRIDE 520    // halfs per LDS row for K=512 (8-half pad -> conflict-free)
#define LSTRIDE2 264   // halfs per LDS row for K=256

typedef __attribute__((ext_vector_type(16))) __bf16 v16bf;
typedef __attribute__((ext_vector_type(8)))  __bf16 v8bf;
typedef __attribute__((ext_vector_type(8)))  float  v8f;

static __device__ __forceinline__ unsigned short f2bf(float x) {
  unsigned int u = __float_as_uint(x);
  unsigned int r = u + 0x7FFFu + ((u >> 16) & 1u);   // round-to-nearest-even
  return (unsigned short)(r >> 16);
}

static __device__ __forceinline__ v16bf cat16(v8bf lo, v8bf hi) {
  v16bf r;
#pragma unroll
  for (int i = 0; i < 8; ++i) { r[i] = lo[i]; r[i + 8] = hi[i]; }
  return r;
}

// Per-wave (RG*16)xK @ Kx32 -> RG x (16x32) accumulation with
// v_wmma_f32_16x16x32_bf16. Each B fragment pair is reused across RG
// row-groups (register-level reuse cuts weight streaming by RG x).
// Wp layout: [(kt*(DIM/16)+nt)*32 + lane]*16 halfs, contiguous per lane.
template <int KSTEPS, int STRIDE, int RG>
static __device__ __forceinline__ void wmma_tile(const unsigned short* ldsX,
                                                 const unsigned short* Wp,
                                                 int lane, int nbase,
                                                 v8f c[RG][2]) {
  const int arow = lane & 15;
  const int k0 = (lane >> 4) * 8;            // A frag: K halves {k0..k0+7, k0+16..k0+23}
  const int ntiles = DIM / 16;
#pragma unroll
  for (int kt = 0; kt < KSTEPS; ++kt) {
    const unsigned short* wp =
        Wp + ((size_t)(kt * ntiles + (nbase >> 4)) * 32 + lane) * 16;
    v16bf b0 = *(const v16bf*)(wp);
    v16bf b1 = *(const v16bf*)(wp + 512);    // next 16-col tile (32 lanes * 16 halfs)
#pragma unroll
    for (int rg = 0; rg < RG; ++rg) {
      const unsigned short* rowp = ldsX + (rg * 16 + arow) * STRIDE;
      v8bf lo = *(const v8bf*)(rowp + kt * 32 + k0);
      v8bf hi = *(const v8bf*)(rowp + kt * 32 + k0 + 16);
      v16bf a = cat16(lo, hi);
      c[rg][0] = __builtin_amdgcn_wmma_f32_16x16x32_bf16(false, a, false, b0, (short)0, c[rg][0], false, false);
      c[rg][1] = __builtin_amdgcn_wmma_f32_16x16x32_bf16(false, a, false, b1, (short)0, c[rg][1], false, false);
    }
  }
}

// ---------- fused gather + edge-MLP + segment-sum (atomics), 64-edge tiles ----------
__global__ __launch_bounds__(256)
void edge_msg_kernel(const float* __restrict__ featA, const int* __restrict__ idxA,
                     const float* __restrict__ featB, const int* __restrict__ idxB,
                     const unsigned short* __restrict__ Wp, const float* __restrict__ bias,
                     const int* __restrict__ outIdx, float* __restrict__ aggr, int E) {
  __shared__ __align__(16) unsigned short ldsX[64 * LSTRIDE];
  __shared__ int ldsDst[64];
  const int tid = threadIdx.x;
  const int lane = tid & 31;
  const int wave = tid >> 5;
  const int e0 = blockIdx.x * 64;

  if (tid < 64) ldsDst[tid] = (e0 + tid < E) ? outIdx[e0 + tid] : -1;
  {
    const int r = tid >> 2;          // 4 threads per row
    const int seg = tid & 3;
    int e = e0 + r; if (e >= E) e = E - 1;
    const int h0 = seg * 128;        // 128-half chunk stays within one source half
    const float* src = (h0 < DIM) ? (featA + (size_t)idxA[e] * DIM + h0)
                                  : (featB + (size_t)idxB[e] * DIM + (h0 - DIM));
    unsigned short* d = &ldsX[r * LSTRIDE + h0];
#pragma unroll 8
    for (int i = 0; i < 128; i += 4) {
      float4 v = *(const float4*)(src + i);
      d[i + 0] = f2bf(v.x); d[i + 1] = f2bf(v.y);
      d[i + 2] = f2bf(v.z); d[i + 3] = f2bf(v.w);
    }
  }
  __syncthreads();

  v8f c[4][2];
  v8f zero = {};
#pragma unroll
  for (int rg = 0; rg < 4; ++rg) { c[rg][0] = zero; c[rg][1] = zero; }
  const int nbase = wave * 32;
  wmma_tile<16, LSTRIDE, 4>(ldsX, Wp, lane, nbase, c);

  const int col = nbase + (lane & 15);
  const int rb = (lane >> 4) * 8;
  const float b0 = bias[col], b1 = bias[col + 16];
#pragma unroll
  for (int rg = 0; rg < 4; ++rg) {
#pragma unroll
    for (int v = 0; v < 8; ++v) {
      int node = ldsDst[rg * 16 + rb + v];
      if (node >= 0) {
        float x0 = fmaxf(c[rg][0][v] + b0, 0.f);
        float x1 = fmaxf(c[rg][1][v] + b1, 0.f);
        atomicAdd(&aggr[(size_t)node * DIM + col], x0);
        atomicAdd(&aggr[(size_t)node * DIM + col + 16], x1);
      }
    }
  }
}

// ---------- combine MLP: out = relu([self|aggr]@W + b) (+ residual), 64-row tiles ----------
__global__ __launch_bounds__(256)
void combine_kernel(const float* __restrict__ selfF, const float* __restrict__ aggrF,
                    const unsigned short* __restrict__ Wp, const float* __restrict__ bias,
                    const float* __restrict__ residual, float* __restrict__ out, int M) {
  __shared__ __align__(16) unsigned short ldsX[64 * LSTRIDE];
  const int tid = threadIdx.x;
  const int lane = tid & 31;
  const int wave = tid >> 5;
  const int m0 = blockIdx.x * 64;
  {
    const int r = tid >> 2;
    const int seg = tid & 3;
    int m = m0 + r; if (m >= M) m = M - 1;
    const int h0 = seg * 128;
    const float* src = (h0 < DIM) ? (selfF + (size_t)m * DIM + h0)
                                  : (aggrF + (size_t)m * DIM + (h0 - DIM));
    unsigned short* d = &ldsX[r * LSTRIDE + h0];
#pragma unroll 8
    for (int i = 0; i < 128; i += 4) {
      float4 v = *(const float4*)(src + i);
      d[i + 0] = f2bf(v.x); d[i + 1] = f2bf(v.y);
      d[i + 2] = f2bf(v.z); d[i + 3] = f2bf(v.w);
    }
  }
  __syncthreads();

  v8f c[4][2];
  v8f zero = {};
#pragma unroll
  for (int rg = 0; rg < 4; ++rg) { c[rg][0] = zero; c[rg][1] = zero; }
  const int nbase = wave * 32;
  wmma_tile<16, LSTRIDE, 4>(ldsX, Wp, lane, nbase, c);

  const int col = nbase + (lane & 15);
  const int rb = (lane >> 4) * 8;
  const float b0 = bias[col], b1 = bias[col + 16];
#pragma unroll
  for (int rg = 0; rg < 4; ++rg) {
#pragma unroll
    for (int v = 0; v < 8; ++v) {
      int m = m0 + rg * 16 + rb + v;
      if (m < M) {
        float x0 = fmaxf(c[rg][0][v] + b0, 0.f);
        float x1 = fmaxf(c[rg][1][v] + b1, 0.f);
        if (residual) {
          x0 += residual[(size_t)m * DIM + col];
          x1 += residual[(size_t)m * DIM + col + 16];
        }
        out[(size_t)m * DIM + col] = x0;
        out[(size_t)m * DIM + col + 16] = x1;
      }
    }
  }
}

// ---------- attention projection + alpha-weighted segment sum (K=256) ----------
__global__ __launch_bounds__(256)
void att_aggr_kernel(const float* __restrict__ factors, const unsigned short* __restrict__ Wp,
                     const float* __restrict__ bias, const float* __restrict__ alpha,
                     const int* __restrict__ batch, float* __restrict__ g_agg, int M) {
  __shared__ __align__(16) unsigned short ldsX[16 * LSTRIDE2];
  __shared__ float ldsAlpha[16];
  __shared__ int ldsBatch[16];
  const int tid = threadIdx.x;
  const int lane = tid & 31;
  const int wave = tid >> 5;
  const int m0 = blockIdx.x * 16;
  if (tid < 16) {
    int m = m0 + tid;
    ldsAlpha[tid] = (m < M) ? alpha[m] : 0.f;
    ldsBatch[tid] = (m < M) ? batch[m] : 0;
  }
  {
    const int r = tid >> 4;
    const int seg = tid & 15;
    int m = m0 + r; if (m >= M) m = M - 1;
    const int h0 = seg * 16;
    const float* src = factors + (size_t)m * DIM + h0;
    unsigned short* d = &ldsX[r * LSTRIDE2 + h0];
#pragma unroll
    for (int i = 0; i < 16; i += 4) {
      float4 v = *(const float4*)(src + i);
      d[i + 0] = f2bf(v.x); d[i + 1] = f2bf(v.y);
      d[i + 2] = f2bf(v.z); d[i + 3] = f2bf(v.w);
    }
  }
  __syncthreads();

  v8f c[1][2];
  v8f zero = {};
  c[0][0] = zero; c[0][1] = zero;
  const int nbase = wave * 32;
  wmma_tile<8, LSTRIDE2, 1>(ldsX, Wp, lane, nbase, c);

  const int col = nbase + (lane & 15);
  const int rb = (lane >> 4) * 8;
  const float b0 = bias[col], b1 = bias[col + 16];
#pragma unroll
  for (int v = 0; v < 8; ++v) {
    int r = rb + v;
    if (m0 + r < M) {
      float a = ldsAlpha[r];
      int g = ldsBatch[r];
      atomicAdd(&g_agg[(size_t)g * DIM + col], (c[0][0][v] + b0) * a);
      atomicAdd(&g_agg[(size_t)g * DIM + col + 16], (c[0][1][v] + b1) * a);
    }
  }
}

// ---------- tiny global-node MLP: out = relu(X@W + b), K=256 ----------
__global__ __launch_bounds__(256)
void gmlp_kernel(const float* __restrict__ X, const unsigned short* __restrict__ Wp,
                 const float* __restrict__ bias, float* __restrict__ out, int M) {
  __shared__ __align__(16) unsigned short ldsX[16 * LSTRIDE2];
  const int tid = threadIdx.x;
  const int lane = tid & 31;
  const int wave = tid >> 5;
  const int m0 = blockIdx.x * 16;
  {
    const int r = tid >> 4;
    const int seg = tid & 15;
    int m = m0 + r; if (m >= M) m = M - 1;
    const int h0 = seg * 16;
    const float* src = X + (size_t)m * DIM + h0;
    unsigned short* d = &ldsX[r * LSTRIDE2 + h0];
#pragma unroll
    for (int i = 0; i < 16; i += 4) {
      float4 v = *(const float4*)(src + i);
      d[i + 0] = f2bf(v.x); d[i + 1] = f2bf(v.y);
      d[i + 2] = f2bf(v.z); d[i + 3] = f2bf(v.w);
    }
  }
  __syncthreads();

  v8f c[1][2];
  v8f zero = {};
  c[0][0] = zero; c[0][1] = zero;
  const int nbase = wave * 32;
  wmma_tile<8, LSTRIDE2, 1>(ldsX, Wp, lane, nbase, c);

  const int col = nbase + (lane & 15);
  const int rb = (lane >> 4) * 8;
  const float b0 = bias[col], b1 = bias[col + 16];
#pragma unroll
  for (int v = 0; v < 8; ++v) {
    int m = m0 + rb + v;
    if (m < M) {
      out[(size_t)m * DIM + col] = fmaxf(c[0][0][v] + b0, 0.f);
      out[(size_t)m * DIM + col + 16] = fmaxf(c[0][1][v] + b1, 0.f);
    }
  }
}

// ---------- weight pack: fp32 [K][N] -> bf16 WMMA-B fragment order ----------
__global__ __launch_bounds__(256)
void pack_kernel(const float* __restrict__ W, unsigned short* __restrict__ Wp,
                 int K, int N) {
  int idx = blockIdx.x * 256 + threadIdx.x;
  if (idx >= K * N) return;
  int h = idx & 15;
  int lane = (idx >> 4) & 31;
  int t = idx >> 9;
  int ntiles = N / 16;
  int nt = t % ntiles;
  int kt = t / ntiles;
  int k = kt * 32 + (lane >> 4) * 8 + (h & 7) + (h >> 3) * 16;   // B 32x16 bf16: lane<16 -> K 0..7,16..23, lane>=16 -> K 8..15,24..31
  int n = nt * 16 + (lane & 15);
  Wp[idx] = f2bf(W[(size_t)k * N + n]);
}

// ---------- small helper kernels ----------
__global__ void fill_kernel(float* p, float v, size_t n) {
  size_t i = (size_t)blockIdx.x * 256 + threadIdx.x;
  if (i < n) p[i] = v;
}

__global__ __launch_bounds__(256)
void gate_kernel(const float* __restrict__ factors, const float* __restrict__ gW,
                 const float* __restrict__ gb, float* __restrict__ gate, int M) {
  int lane = threadIdx.x & 31;
  int row = blockIdx.x * 8 + (threadIdx.x >> 5);
  if (row >= M) return;
  const float* f = factors + (size_t)row * DIM;
  float s = 0.f;
  for (int i = lane; i < DIM; i += 32) s += f[i] * gW[i];
#pragma unroll
  for (int off = 16; off > 0; off >>= 1) s += __shfl_down(s, off, 32);
  if (lane == 0) gate[row] = s + gb[0];
}

static __device__ void atomicMaxFloat(float* addr, float val) {
  unsigned int* ua = (unsigned int*)addr;
  unsigned int old = __float_as_uint(*addr);
  while (__uint_as_float(old) < val) {
    unsigned int prev = atomicCAS(ua, old, __float_as_uint(val));
    if (prev == old) break;
    old = prev;
  }
}

__global__ void segmax_kernel(const float* gate, const int* batch, float* mx, int M) {
  int i = blockIdx.x * 256 + threadIdx.x;
  if (i < M) atomicMaxFloat(&mx[batch[i]], gate[i]);
}

__global__ void expsum_kernel(const float* gate, const int* batch, const float* mx,
                              float* eexp, float* denom, int M) {
  int i = blockIdx.x * 256 + threadIdx.x;
  if (i < M) {
    float e = expf(gate[i] - mx[batch[i]]);
    eexp[i] = e;
    atomicAdd(&denom[batch[i]], e);
  }
}

__global__ void alpha_kernel(float* eexp, const int* batch, const float* denom, int M) {
  int i = blockIdx.x * 256 + threadIdx.x;
  if (i < M) eexp[i] = eexp[i] / denom[batch[i]];
}

extern "C" void kernel_launch(void* const* d_in, const int* in_sizes, int n_in,
                              void* d_out, int out_size, void* d_ws, size_t ws_size,
                              hipStream_t stream) {
  const float* variables = (const float*)d_in[0];
  const float* factors   = (const float*)d_in[1];
  /* d_in[2] = edge_attr: carried but unused by the math */
  const int*   edge_index = (const int*)d_in[3];
  const int*   batch_idx  = (const int*)d_in[4];
  const float* mW_v2f = (const float*)d_in[5];
  const float* mb_v2f = (const float*)d_in[6];
  const float* cW_v2f = (const float*)d_in[7];
  const float* cb_v2f = (const float*)d_in[8];
  const float* mW_f2v = (const float*)d_in[9];
  const float* mb_f2v = (const float*)d_in[10];
  const float* cW_f2v = (const float*)d_in[11];
  const float* cb_f2v = (const float*)d_in[12];
  const float* gate_W = (const float*)d_in[13];
  const float* gate_b = (const float*)d_in[14];
  const float* att_W  = (const float*)d_in[15];
  const float* att_b  = (const float*)d_in[16];
  const float* gl_W   = (const float*)d_in[17];
  const float* gl_b   = (const float*)d_in[18];

  const int NV = in_sizes[0] / DIM;
  const int NF = in_sizes[1] / DIM;
  const int E  = in_sizes[2];
  const int L  = in_sizes[6] / DIM;          // mb_v2f is (L, D)
  const int* srcIdx = edge_index;            // row 0: variable index
  const int* dstIdx = edge_index + E;        // row 1: factor index
  const size_t KN = (size_t)2 * DIM * DIM;   // one layer's 2D x D weight

  // workspace carve (256-byte aligned)
  char* w = (char*)d_ws;
  auto take = [&](size_t bytes) -> char* {
    char* p = w; w += (bytes + 255) & ~(size_t)255; return p;
  };
  float* factWs = (float*)take((size_t)NF * DIM * 4);
  float* varWs  = (float*)take((size_t)NV * DIM * 4);
  float* aggr   = (float*)take((size_t)NF * DIM * 4);   // shared F/V aggregation buffer
  unsigned short* p_mW_v2f = (unsigned short*)take((size_t)L * KN * 2);
  unsigned short* p_cW_v2f = (unsigned short*)take((size_t)L * KN * 2);
  unsigned short* p_mW_f2v = (unsigned short*)take((size_t)L * KN * 2);
  unsigned short* p_cW_f2v = (unsigned short*)take((size_t)L * KN * 2);
  unsigned short* p_att    = (unsigned short*)take((size_t)DIM * DIM * 2);
  unsigned short* p_gl     = (unsigned short*)take((size_t)DIM * DIM * 2);
  float* gate  = (float*)take((size_t)NF * 4);
  float* eexp  = (float*)take((size_t)NF * 4);
  float* mx    = (float*)take(GNUM * 4);
  float* denom = (float*)take(GNUM * 4);
  float* g_agg = (float*)take((size_t)GNUM * DIM * 4);

  // pack weights to bf16 fragment order
  const int pb = (int)((KN + 255) / 256);
  for (int l = 0; l < L; ++l) {
    pack_kernel<<<pb, 256, 0, stream>>>(mW_v2f + l * KN, p_mW_v2f + l * KN, 2 * DIM, DIM);
    pack_kernel<<<pb, 256, 0, stream>>>(cW_v2f + l * KN, p_cW_v2f + l * KN, 2 * DIM, DIM);
    pack_kernel<<<pb, 256, 0, stream>>>(mW_f2v + l * KN, p_mW_f2v + l * KN, 2 * DIM, DIM);
    pack_kernel<<<pb, 256, 0, stream>>>(cW_f2v + l * KN, p_cW_f2v + l * KN, 2 * DIM, DIM);
  }
  pack_kernel<<<(DIM * DIM + 255) / 256, 256, 0, stream>>>(att_W, p_att, DIM, DIM);
  // gl_W is (2D, D); g_prev = 0 so only the first D rows matter
  pack_kernel<<<(DIM * DIM + 255) / 256, 256, 0, stream>>>(gl_W, p_gl, DIM, DIM);

  float* vOutFinal = (float*)d_out;
  float* fOutFinal = (float*)d_out + (size_t)NV * DIM;
  float* gOut      = (float*)d_out + (size_t)(NV + NF) * DIM;

  const float* vIn = variables;
  const float* fIn = factors;
  const int eBlocks = (E + 63) / 64;
  for (int l = 0; l < L; ++l) {
    float* fOut = (l == L - 1) ? fOutFinal : factWs;
    float* vOut = (l == L - 1) ? vOutFinal : varWs;
    // variable -> factor: msg = relu([F[dst] | V[src]] @ mW), aggr over dst
    fill_kernel<<<(int)(((size_t)NF * DIM + 255) / 256), 256, 0, stream>>>(aggr, 0.f, (size_t)NF * DIM);
    edge_msg_kernel<<<eBlocks, 256, 0, stream>>>(fIn, dstIdx, vIn, srcIdx,
                                                 p_mW_v2f + l * KN, mb_v2f + l * DIM,
                                                 dstIdx, aggr, E);
    combine_kernel<<<(NF + 63) / 64, 256, 0, stream>>>(fIn, aggr, p_cW_v2f + l * KN,
                                                       cb_v2f + l * DIM, nullptr, fOut, NF);
    // factor -> variable: msg = relu([V[src] | F[dst]] @ mW), aggr over src; residual add
    fill_kernel<<<(int)(((size_t)NV * DIM + 255) / 256), 256, 0, stream>>>(aggr, 0.f, (size_t)NV * DIM);
    edge_msg_kernel<<<eBlocks, 256, 0, stream>>>(vIn, srcIdx, fOut, dstIdx,
                                                 p_mW_f2v + l * KN, mb_f2v + l * DIM,
                                                 srcIdx, aggr, E);
    combine_kernel<<<(NV + 63) / 64, 256, 0, stream>>>(vIn, aggr, p_cW_f2v + l * KN,
                                                       cb_f2v + l * DIM, vIn, vOut, NV);
    fIn = fOut; vIn = vOut;
  }

  // GlobalNode: attentional aggregation over factors per graph
  gate_kernel<<<(NF + 7) / 8, 256, 0, stream>>>(fIn, gate_W, gate_b, gate, NF);
  fill_kernel<<<1, 256, 0, stream>>>(mx, -3.402823466e38f, GNUM);
  fill_kernel<<<1, 256, 0, stream>>>(denom, 0.f, GNUM);
  fill_kernel<<<(GNUM * DIM + 255) / 256, 256, 0, stream>>>(g_agg, 0.f, (size_t)GNUM * DIM);
  segmax_kernel<<<(NF + 255) / 256, 256, 0, stream>>>(gate, batch_idx, mx, NF);
  expsum_kernel<<<(NF + 255) / 256, 256, 0, stream>>>(gate, batch_idx, mx, eexp, denom, NF);
  alpha_kernel<<<(NF + 255) / 256, 256, 0, stream>>>(eexp, batch_idx, denom, NF);
  att_aggr_kernel<<<(NF + 15) / 16, 256, 0, stream>>>(fIn, p_att, att_b, eexp,
                                                      batch_idx, g_agg, NF);
  gmlp_kernel<<<(GNUM + 15) / 16, 256, 0, stream>>>(g_agg, p_gl, gl_b, gOut, GNUM);
}
